// Block_ViT_42125039239289
// MI455X (gfx1250) — matrix operand order repaired and checked by reading, not compiled, not run on hardware
//
#include <hip/hip_runtime.h>
#include <hip/hip_bf16.h>

// ---------------------------------------------------------------------------
// Block_ViT on MI455X (gfx1250): all GEMMs via v_wmma_f32_16x16x32_f16.
// f32 activations/weights are packed once into CDNA5 WMMA fragment layouts
// (f16), GEMM accumulates in f32.  Each wave computes a 32x32 macro-tile
// (2x2 WMMA tiles): 4 wmma per K-step off 4 fragment loads -> ~16 FLOP/byte.
// Needs ~480 MB workspace.
// ---------------------------------------------------------------------------

typedef __attribute__((ext_vector_type(16))) _Float16 v16h;
typedef __attribute__((ext_vector_type(8)))  float    v8f;

#define NSP 4096      // 64*64 spatial
#define NB  8192      // B * NSP, B = 2
#define HEADS 4
#define KVC 960

__device__ __forceinline__ float gelu_f(float x) {
    return 0.5f * x * (1.f + erff(x * 0.70710678118654752f));
}

// A-fragment offset: 16(M) x 32(K) f16 tile, per CDNA5 ISA layout.
// lane = (m&15) + 16*((kr>>3)&1), half = (kr&7) + 8*((kr>>4)&1)
__device__ __forceinline__ size_t paOff(int m, int k, int Kt) {
    int mt = m >> 4, kt = k >> 5, kr = k & 31;
    int lane = (m & 15) + 16 * ((kr >> 3) & 1);
    int h    = (kr & 7) + 8 * ((kr >> 4) & 1);
    return ((size_t)(mt * (Kt >> 5) + kt) * 32 + lane) * 16 + h;
}

// B-fragment offset: 32(K) x 16(N) f16 tile.
// lane = (j&15) + 16*((kr>>3)&1), half = (kr&7) + 8*((kr>>4)&1)
__device__ __forceinline__ size_t pbOff(int k, int j, int Nt) {
    int kt = k >> 5, kr = k & 31, jt = j >> 4;
    int lane = (j & 15) + 16 * ((kr >> 3) & 1);
    int h    = (kr & 7) + 8 * ((kr >> 4) & 1);
    return ((size_t)(kt * (Nt >> 4) + jt) * 32 + lane) * 16 + h;
}

// ---------------------------------------------------------------------------
// Core WMMA GEMM: D[Mt,Nt] = A[Mt,Kt] * B[Kt,Nt]  (f16 in, f32 out)
// 32x32 macro-tile per wave (2x2 16x16 tiles, 4 independent accumulators).
// Requires Mt % 32 == 0, Nt % 32 == 0, Kt % 32 == 0 (true for every call).
// EPI: 0 = none, 1 = +bias, 2 = +bias then exact GELU
// ---------------------------------------------------------------------------
template <int EPI>
__global__ void gemm_wmma(const _Float16* __restrict__ A,
                          const _Float16* __restrict__ Bm,
                          float* __restrict__ D,
                          const float* __restrict__ bias,
                          int Mt, int Nt, int Kt) {
    int lane = threadIdx.x & 31;
    int wave = threadIdx.x >> 5;
    int ntN  = Nt >> 4;                       // 16-wide N tiles
    int nMac = ntN >> 1;                      // 32-wide macro columns
    int tile = blockIdx.x * 4 + wave;
    int numMac = (Mt >> 5) * nMac;
    if (tile >= numMac) return;               // wave-uniform: EXEC all-ones
    int mm = tile / nMac, nn = tile % nMac;   // macro-tile coords
    int mt0 = mm * 2, nt0 = nn * 2;
    int kTiles = Kt >> 5;
    v8f acc00 = {}, acc01 = {}, acc10 = {}, acc11 = {};
    const _Float16* aBase0 = A  + ((size_t)mt0 * kTiles * 32 + lane) * 16;
    const _Float16* aBase1 = aBase0 + (size_t)kTiles * 512;
    const _Float16* bBase0 = Bm + ((size_t)nt0 * 32 + lane) * 16;
    const _Float16* bBase1 = bBase0 + 512;
    size_t bStep = (size_t)ntN * 512;
    for (int kt = 0; kt < kTiles; ++kt) {
        v16h a0 = *(const v16h*)(aBase0 + (size_t)kt * 512);
        v16h a1 = *(const v16h*)(aBase1 + (size_t)kt * 512);
        v16h b0 = *(const v16h*)(bBase0 + (size_t)kt * bStep);
        v16h b1 = *(const v16h*)(bBase1 + (size_t)kt * bStep);
        acc00 = __builtin_amdgcn_wmma_f32_16x16x32_f16(false, a0, false, b0,
                                                       (short)0, acc00, false, false);
        acc01 = __builtin_amdgcn_wmma_f32_16x16x32_f16(false, a0, false, b1,
                                                       (short)0, acc01, false, false);
        acc10 = __builtin_amdgcn_wmma_f32_16x16x32_f16(false, a1, false, b0,
                                                       (short)0, acc10, false, false);
        acc11 = __builtin_amdgcn_wmma_f32_16x16x32_f16(false, a1, false, b1,
                                                       (short)0, acc11, false, false);
    }
    int colL  = lane & 15;
    int rOff  = 8 * (lane >> 4);
    v8f accs[4] = {acc00, acc01, acc10, acc11};
#pragma unroll
    for (int im = 0; im < 2; ++im) {
#pragma unroll
        for (int in = 0; in < 2; ++in) {
            v8f acc = accs[im * 2 + in];
            int rbase = (mt0 + im) * 16 + rOff;
            int col   = (nt0 + in) * 16 + colL;
#pragma unroll
            for (int g = 0; g < 8; ++g) {
                int row = rbase + g;
                float v = acc[g];
                if (EPI >= 1) v += bias[row];
                if (EPI == 2) v = gelu_f(v);
                D[(size_t)row * Nt + col] = v;
            }
        }
    }
}

// ---------------------------------------------------------------------------
// Fragment packers (f32 strided source -> f16 fragments)
// ---------------------------------------------------------------------------
__global__ void pack_A_k(const float* __restrict__ src, long long rs, long long cs,
                         int Mt, int Kt, _Float16* __restrict__ dst) {
    long long i = (long long)blockIdx.x * 256 + threadIdx.x;
    if (i >= (long long)Mt * Kt) return;
    int m = (int)(i / Kt), k = (int)(i % Kt);
    dst[paOff(m, k, Kt)] = (_Float16)src[(long long)m * rs + (long long)k * cs];
}

__global__ void pack_B_k(const float* __restrict__ src, long long rs, long long cs,
                         int Kt, int Nt, _Float16* __restrict__ dst) {
    long long i = (long long)blockIdx.x * 256 + threadIdx.x;
    if (i >= (long long)Kt * Nt) return;
    int k = (int)(i / Nt), j = (int)(i % Nt);
    dst[pbOff(k, j, Nt)] = (_Float16)src[(long long)k * rs + (long long)j * cs];
}

// ---------------------------------------------------------------------------
// LayerNorm over 960 concatenated channels -> packed B (K=960, cols b-major)
// one block (256 thr) per column j in [0, 8192)
// ---------------------------------------------------------------------------
__global__ void ln_all_packB(const float* __restrict__ e1, const float* __restrict__ e2,
                             const float* __restrict__ e3, const float* __restrict__ e4,
                             const float* __restrict__ w, const float* __restrict__ bb,
                             _Float16* __restrict__ dst) {
    int j = blockIdx.x;
    int b = j >> 12, n = j & (NSP - 1);
    __shared__ float r1[256], r2[256];
    float s = 0.f, ss = 0.f;
    for (int c = threadIdx.x; c < KVC; c += 256) {
        const float* src; int cc; int ct;
        if (c < 64)       { src = e1; cc = c;       ct = 64;  }
        else if (c < 192) { src = e2; cc = c - 64;  ct = 128; }
        else if (c < 448) { src = e3; cc = c - 192; ct = 256; }
        else              { src = e4; cc = c - 448; ct = 512; }
        float v = src[((size_t)b * ct + cc) * NSP + n];
        s += v; ss += v * v;
    }
    r1[threadIdx.x] = s; r2[threadIdx.x] = ss; __syncthreads();
    for (int o = 128; o > 0; o >>= 1) {
        if (threadIdx.x < o) { r1[threadIdx.x] += r1[threadIdx.x + o];
                               r2[threadIdx.x] += r2[threadIdx.x + o]; }
        __syncthreads();
    }
    float mu = r1[0] / KVC;
    float rs = rsqrtf(fmaxf(r2[0] / KVC - mu * mu, 0.f) + 1e-5f);
    for (int c = threadIdx.x; c < KVC; c += 256) {
        const float* src; int cc; int ct;
        if (c < 64)       { src = e1; cc = c;       ct = 64;  }
        else if (c < 192) { src = e2; cc = c - 64;  ct = 128; }
        else if (c < 448) { src = e3; cc = c - 192; ct = 256; }
        else              { src = e4; cc = c - 448; ct = 512; }
        float v = src[((size_t)b * ct + cc) * NSP + n];
        dst[pbOff(c, j, NB)] = (_Float16)((v - mu) * rs * w[c] + bb[c]);
    }
}

// Per-branch channel LayerNorm; optional f32 out, optional packed-B out.
template <bool WF32, bool PACK>
__global__ void ln_branch(const float* __restrict__ src, int C,
                          const float* __restrict__ w, const float* __restrict__ bb,
                          float* __restrict__ xout, _Float16* __restrict__ packOut) {
    int j = blockIdx.x;
    int b = j >> 12, n = j & (NSP - 1);
    __shared__ float r1[256], r2[256];
    float s = 0.f, ss = 0.f;
    for (int c = threadIdx.x; c < C; c += 256) {
        float v = src[((size_t)b * C + c) * NSP + n];
        s += v; ss += v * v;
    }
    r1[threadIdx.x] = s; r2[threadIdx.x] = ss; __syncthreads();
    for (int o = 128; o > 0; o >>= 1) {
        if (threadIdx.x < o) { r1[threadIdx.x] += r1[threadIdx.x + o];
                               r2[threadIdx.x] += r2[threadIdx.x + o]; }
        __syncthreads();
    }
    float mu = r1[0] / C;
    float rs = rsqrtf(fmaxf(r2[0] / C - mu * mu, 0.f) + 1e-5f);
    for (int c = threadIdx.x; c < C; c += 256) {
        float v = (src[((size_t)b * C + c) * NSP + n] - mu) * rs * w[c] + bb[c];
        if (WF32) xout[((size_t)b * C + c) * NSP + n] = v;
        if (PACK) packOut[pbOff(c, j, NB)] = (_Float16)v;
    }
}

// ---------------------------------------------------------------------------
// Depthwise conv over (ch, b, 64, 64) layout (chStride = 8192, bStride = 4096)
// ---------------------------------------------------------------------------
template <int KS, bool GELU, bool MUL>
__global__ void dwconv(const float* __restrict__ src, float* __restrict__ dst,
                       const float* __restrict__ w, const float* __restrict__ bias,
                       const float* __restrict__ mul, int C) {
    size_t i = (size_t)blockIdx.x * 256 + threadIdx.x;
    if (i >= (size_t)C * NB) return;
    int n = i & (NSP - 1); int b = (i >> 12) & 1; int ch = (int)(i >> 13);
    int y = n >> 6, x = n & 63;
    const float* sp = src + (size_t)ch * NB + (size_t)b * NSP;
    const float* wc = w + (size_t)ch * KS * KS;
    float acc = bias ? bias[ch] : 0.f;
    const int P = KS / 2;
#pragma unroll
    for (int ky = 0; ky < KS; ++ky) {
        int yy = y + ky - P; if (yy < 0 || yy > 63) continue;
#pragma unroll
        for (int kx = 0; kx < KS; ++kx) {
            int xx = x + kx - P; if (xx < 0 || xx > 63) continue;
            acc += wc[ky * KS + kx] * sp[yy * 64 + xx];
        }
    }
    if (GELU) acc = gelu_f(acc);
    if (MUL)  acc *= mul[i];
    dst[i] = acc;
}

// Grouped conv for q: out ch o uses in channels {2*(o/2), 2*(o/2)+1}, 3x3 pad1
__global__ void qconv(const float* __restrict__ src, float* __restrict__ dst,
                      const float* __restrict__ w, int O) {
    size_t i = (size_t)blockIdx.x * 256 + threadIdx.x;
    if (i >= (size_t)O * NB) return;
    int n = i & (NSP - 1); int b = (i >> 12) & 1; int o = (int)(i >> 13);
    int y = n >> 6, x = n & 63;
    int g = o >> 1;
    float acc = 0.f;
#pragma unroll
    for (int ic = 0; ic < 2; ++ic) {
        const float* sp = src + ((size_t)(2 * g + ic)) * NB + (size_t)b * NSP;
        const float* wc = w + ((size_t)o * 2 + ic) * 9;
#pragma unroll
        for (int ky = 0; ky < 3; ++ky) {
            int yy = y + ky - 1; if (yy < 0 || yy > 63) continue;
#pragma unroll
            for (int kx = 0; kx < 3; ++kx) {
                int xx = x + kx - 1; if (xx < 0 || xx > 63) continue;
                acc += wc[ky * 3 + kx] * sp[yy * 64 + xx];
            }
        }
    }
    dst[i] = acc;
}

// Dense partial conv (dc -> dc, 3x3 pad1) on first dc channels of (b,C,n) x.
__global__ void pconv(const float* __restrict__ xln, float* __restrict__ dst,
                      const float* __restrict__ w, int C, int dc) {
    size_t i = (size_t)blockIdx.x * 256 + threadIdx.x;
    if (i >= (size_t)2 * dc * NSP) return;
    int n = i & (NSP - 1); int bo = (int)(i >> 12);
    int b = bo / dc, o = bo % dc;
    int y = n >> 6, x = n & 63;
    float acc = 0.f;
    for (int ic = 0; ic < dc; ++ic) {
        const float* sp = xln + ((size_t)b * C + ic) * NSP;
        const float* wc = w + ((size_t)o * dc + ic) * 9;
#pragma unroll
        for (int ky = 0; ky < 3; ++ky) {
            int yy = y + ky - 1; if (yy < 0 || yy > 63) continue;
#pragma unroll
            for (int kx = 0; kx < 3; ++kx) {
                int xx = x + kx - 1; if (xx < 0 || xx > 63) continue;
                acc += wc[ky * 3 + kx] * sp[yy * 64 + xx];
            }
        }
    }
    dst[i] = acc;
}

// Pack FFN input: first dc channels from pconv out x1 (b,dc,n), rest from xln.
__global__ void packB_ffn(const float* __restrict__ xln, const float* __restrict__ x1,
                          int C, int dc, _Float16* __restrict__ dst) {
    size_t i = (size_t)blockIdx.x * 256 + threadIdx.x;
    if (i >= (size_t)C * NB) return;
    int j = (int)(i % NB); int c = (int)(i / NB);
    int b = j >> 12, n = j & (NSP - 1);
    float v = (c < dc) ? x1[((size_t)b * dc + c) * NSP + n]
                       : xln[((size_t)b * C + c) * NSP + n];
    dst[pbOff(c, j, NB)] = (_Float16)v;
}

// In-place L2 normalize contiguous rows of 4096.
__global__ void l2norm_rows(float* __restrict__ p) {
    float* row = p + (size_t)blockIdx.x * NSP;
    __shared__ float red[256];
    float s = 0.f;
    for (int i = threadIdx.x; i < NSP; i += 256) { float v = row[i]; s += v * v; }
    red[threadIdx.x] = s; __syncthreads();
    for (int o = 128; o > 0; o >>= 1) {
        if (threadIdx.x < o) red[threadIdx.x] += red[threadIdx.x + o];
        __syncthreads();
    }
    float sc = 1.f / fmaxf(sqrtf(red[0]), 1e-12f);
    for (int i = threadIdx.x; i < NSP; i += 256) row[i] *= sc;
}

// Instance-norm stats per (b,h) over (c x 960).  (attn scale cancels here.)
__global__ void inorm_stats(const float* __restrict__ attn, float* __restrict__ stats, int c) {
    int bh = blockIdx.x;
    const float* p = attn + (size_t)bh * c * KVC;
    int n = c * KVC;
    __shared__ float r1[256], r2[256];
    float s = 0.f, ss = 0.f;
    for (int i = threadIdx.x; i < n; i += 256) { float v = p[i]; s += v; ss += v * v; }
    r1[threadIdx.x] = s; r2[threadIdx.x] = ss; __syncthreads();
    for (int o = 128; o > 0; o >>= 1) {
        if (threadIdx.x < o) { r1[threadIdx.x] += r1[threadIdx.x + o];
                               r2[threadIdx.x] += r2[threadIdx.x + o]; }
        __syncthreads();
    }
    if (threadIdx.x == 0) {
        float mu = r1[0] / n;
        stats[2 * bh]     = mu;
        stats[2 * bh + 1] = rsqrtf(fmaxf(r2[0] / n - mu * mu, 0.f) + 1e-5f);
    }
}

// (x-mu)*rs -> softmax over 960 -> write f16 packed-A fragments directly.
__global__ void softmax_pack(const float* __restrict__ attn, const float* __restrict__ stats,
                             _Float16* __restrict__ prA, int c) {
    int gr = blockIdx.x;                 // [0, 8*c)
    int bh = gr / c, cc = gr % c;
    const float* row = attn + (size_t)bh * c * KVC + (size_t)cc * KVC;
    float mu = stats[2 * bh], rs = stats[2 * bh + 1];
    __shared__ float sh[KVC];
    __shared__ float red[256];
    float mx = -3.4e38f;
    for (int i = threadIdx.x; i < KVC; i += 256) {
        float v = (row[i] - mu) * rs; sh[i] = v; mx = fmaxf(mx, v);
    }
    red[threadIdx.x] = mx; __syncthreads();
    for (int o = 128; o > 0; o >>= 1) {
        if (threadIdx.x < o) red[threadIdx.x] = fmaxf(red[threadIdx.x], red[threadIdx.x + o]);
        __syncthreads();
    }
    mx = red[0]; __syncthreads();
    float s = 0.f;
    for (int i = threadIdx.x; i < KVC; i += 256) {
        float e = __expf(sh[i] - mx); sh[i] = e; s += e;
    }
    red[threadIdx.x] = s; __syncthreads();
    for (int o = 128; o > 0; o >>= 1) {
        if (threadIdx.x < o) red[threadIdx.x] += red[threadIdx.x + o];
        __syncthreads();
    }
    float inv = 1.f / red[0];
    _Float16* dst = prA + (size_t)bh * c * KVC;
    for (int i = threadIdx.x; i < KVC; i += 256)
        dst[paOff(cc, i, KVC)] = (_Float16)(sh[i] * inv);
}

// Mean over heads: om[cc*8192 + b*4096 + n] = mean_h obh[((b*4+h)*c+cc)*4096+n]
__global__ void head_mean(const float* __restrict__ obh, float* __restrict__ om, int c) {
    size_t i = (size_t)blockIdx.x * 256 + threadIdx.x;
    if (i >= (size_t)c * NB) return;
    int n = i & (NSP - 1); int b = (i >> 12) & 1; int cc = (int)(i >> 13);
    float s = 0.f;
#pragma unroll
    for (int h = 0; h < HEADS; ++h)
        s += obh[(((size_t)(b * HEADS + h)) * c + cc) * NSP + n];
    om[i] = 0.25f * s;
}

// out[b,c,n] = emb[b,c,n] + oproj[c*8192 + b*4096 + n]
__global__ void residual_add(const float* __restrict__ emb, const float* __restrict__ oproj,
                             float* __restrict__ out, int c) {
    size_t i = (size_t)blockIdx.x * 256 + threadIdx.x;
    if (i >= (size_t)2 * c * NSP) return;
    int n = i & (NSP - 1); size_t bc = i >> 12;
    int cc = (int)(bc % c); int b = (int)(bc / c);
    out[i] = emb[i] + oproj[(size_t)cc * NB + (size_t)b * NSP + n];
}

// Mean over spatial for each (ch,b) row of y (ch,b,n); store g[b*c+ch].
__global__ void col_mean(const float* __restrict__ y, float* __restrict__ g, int c) {
    int r = blockIdx.x;                  // r = ch*2 + b
    int ch = r >> 1, b = r & 1;
    const float* p = y + (size_t)ch * NB + (size_t)b * NSP;
    __shared__ float red[256];
    float s = 0.f;
    for (int i = threadIdx.x; i < NSP; i += 256) s += p[i];
    red[threadIdx.x] = s; __syncthreads();
    for (int o = 128; o > 0; o >>= 1) {
        if (threadIdx.x < o) red[threadIdx.x] += red[threadIdx.x + o];
        __syncthreads();
    }
    if (threadIdx.x == 0) g[b * c + ch] = red[0] / NSP;
}

// ECA: out[b,c,n] += y[(c,b,n)] * sigmoid(conv1d_k3(g)[b,c])
__global__ void eca_add_out(const float* __restrict__ y, const float* __restrict__ g,
                            const float* __restrict__ ew, float* __restrict__ out, int c) {
    size_t i = (size_t)blockIdx.x * 256 + threadIdx.x;
    if (i >= (size_t)2 * c * NSP) return;
    int n = i & (NSP - 1); size_t bc = i >> 12;
    int cc = (int)(bc % c); int b = (int)(bc / c);
    float gg = ew[1] * g[b * c + cc];
    if (cc > 0)     gg += ew[0] * g[b * c + cc - 1];
    if (cc + 1 < c) gg += ew[2] * g[b * c + cc + 1];
    float sg = 1.f / (1.f + __expf(-gg));
    out[i] += y[(size_t)cc * NB + (size_t)b * NSP + n] * sg;
}

// ---------------------------------------------------------------------------
// Host orchestration
// ---------------------------------------------------------------------------
extern "C" void kernel_launch(void* const* d_in, const int* in_sizes, int n_in,
                              void* d_out, int out_size, void* d_ws, size_t ws_size,
                              hipStream_t stream) {
    (void)in_sizes; (void)n_in; (void)out_size; (void)ws_size;
    const float* emb[4] = {(const float*)d_in[0], (const float*)d_in[1],
                           (const float*)d_in[2], (const float*)d_in[3]};
    const int CHv[4] = {64, 128, 256, 512};
    auto prm = [&](int i, int o) { return (const float*)d_in[4 + 17 * i + o]; };
    // param slots per branch: 0 ln_w,1 ln_b,2 mhead_w,3 q_w,4 proj_w,5 ffn_ln_w,
    // 6 ffn_ln_b,7 pconv_w,8 lin1_w,9 lin1_b,10 dw3_w,11 dw3_b,12 dw5_w,13 dw5_b,
    // 14 lin2_w,15 lin2_b,16 eca_w
    const float* ln_all_w = (const float*)d_in[72];
    const float* ln_all_b = (const float*)d_in[73];
    const float* mheadk_w = (const float*)d_in[74];
    const float* mheadv_w = (const float*)d_in[75];
    const float* kdw_w    = (const float*)d_in[76];
    const float* vdw_w    = (const float*)d_in[77];

    char* ws = (char*)d_ws;
    size_t off = 0;
    auto alloc = [&](size_t bytes) -> char* {
        char* p = ws + off; off += (bytes + 255) & ~(size_t)255; return p;
    };
    auto cdiv = [](size_t a, size_t b) { return (unsigned)((a + b - 1) / b); };
    auto gB = [&](int Mt, int Nt) {      // GEMM grid: 4 macro-tiles (32x32)/block
        return cdiv((size_t)(Mt / 32) * (Nt / 32), 4);
    };

    // ---- permanent: packed K (4096x960 per bh) and V (960x4096 per bh) ----
    _Float16* kB = (_Float16*)alloc((size_t)8 * NSP * KVC * 2);
    _Float16* vB = (_Float16*)alloc((size_t)8 * KVC * NSP * 2);
    const size_t scratch0 = off;

    // =================== shared K/V phase ===================
    {
        _Float16* XallB = (_Float16*)alloc((size_t)KVC * NB * 2);
        _Float16* WA    = (_Float16*)alloc((size_t)3840 * KVC * 2);
        float* pre = (float*)alloc((size_t)3840 * NB * 4);
        float* dwb = (float*)alloc((size_t)3840 * NB * 4);

        ln_all_packB<<<NB, 256, 0, stream>>>(emb[0], emb[1], emb[2], emb[3],
                                             ln_all_w, ln_all_b, XallB);
        // K path
        pack_A_k<<<cdiv((size_t)3840 * KVC, 256), 256, 0, stream>>>(
            mheadk_w, KVC, 1, 3840, KVC, WA);
        gemm_wmma<0><<<gB(3840, NB), 128, 0, stream>>>(WA, XallB, pre, nullptr, 3840, NB, KVC);
        dwconv<3, false, false><<<cdiv((size_t)3840 * NB, 256), 256, 0, stream>>>(
            pre, dwb, kdw_w, nullptr, nullptr, 3840);
        l2norm_rows<<<3840 * 2, 256, 0, stream>>>(dwb);
        for (int b = 0; b < 2; ++b)
            for (int h = 0; h < HEADS; ++h) {
                int bh = b * HEADS + h;
                pack_B_k<<<cdiv((size_t)NSP * KVC, 256), 256, 0, stream>>>(
                    dwb + (size_t)h * KVC * NB + (size_t)b * NSP, 1, NB,
                    NSP, KVC, kB + (size_t)bh * NSP * KVC);
            }
        // V path (reuse WA/pre/dwb)
        pack_A_k<<<cdiv((size_t)3840 * KVC, 256), 256, 0, stream>>>(
            mheadv_w, KVC, 1, 3840, KVC, WA);
        gemm_wmma<0><<<gB(3840, NB), 128, 0, stream>>>(WA, XallB, pre, nullptr, 3840, NB, KVC);
        dwconv<3, false, false><<<cdiv((size_t)3840 * NB, 256), 256, 0, stream>>>(
            pre, dwb, vdw_w, nullptr, nullptr, 3840);
        for (int b = 0; b < 2; ++b)
            for (int h = 0; h < HEADS; ++h) {
                int bh = b * HEADS + h;
                pack_B_k<<<cdiv((size_t)KVC * NSP, 256), 256, 0, stream>>>(
                    dwb + (size_t)h * KVC * NB + (size_t)b * NSP, NB, 1,
                    KVC, NSP, vB + (size_t)bh * KVC * NSP);
            }
    }

    // =================== per-branch ===================
    size_t outOff = 0;
    for (int i = 0; i < 4; ++i) {
        const int c = CHv[i];
        float* outp = (float*)d_out + outOff;
        outOff += (size_t)2 * c * NSP;

        // ---- attention phase (scratch reset) ----
        off = scratch0;
        _Float16* xlnB = (_Float16*)alloc((size_t)c * NB * 2);
        _Float16* WqA  = (_Float16*)alloc((size_t)4 * c * c * 2);
        float* qpre = (float*)alloc((size_t)4 * c * NB * 4);
        float* qdw  = (float*)alloc((size_t)4 * c * NB * 4);
        _Float16* qA  = (_Float16*)alloc((size_t)8 * c * NSP * 2);
        float* attn = (float*)alloc((size_t)8 * c * KVC * 4);
        float* stats = (float*)alloc(64);
        _Float16* prA = (_Float16*)alloc((size_t)8 * c * KVC * 2);
        float* obh   = (float*)alloc((size_t)8 * c * NSP * 4);
        float* omean = (float*)alloc((size_t)c * NB * 4);
        _Float16* omB = (_Float16*)alloc((size_t)c * NB * 2);
        _Float16* WpA = (_Float16*)alloc((size_t)c * c * 2);
        float* oproj = (float*)alloc((size_t)c * NB * 4);

        ln_branch<false, true><<<NB, 256, 0, stream>>>(emb[i], c, prm(i, 0), prm(i, 1),
                                                       nullptr, xlnB);
        pack_A_k<<<cdiv((size_t)4 * c * c, 256), 256, 0, stream>>>(
            prm(i, 2), c, 1, 4 * c, c, WqA);
        gemm_wmma<0><<<gB(4 * c, NB), 128, 0, stream>>>(WqA, xlnB, qpre, nullptr, 4 * c, NB, c);
        qconv<<<cdiv((size_t)4 * c * NB, 256), 256, 0, stream>>>(qpre, qdw, prm(i, 3), 4 * c);
        l2norm_rows<<<4 * c * 2, 256, 0, stream>>>(qdw);
        for (int b = 0; b < 2; ++b)
            for (int h = 0; h < HEADS; ++h) {
                int bh = b * HEADS + h;
                pack_A_k<<<cdiv((size_t)c * NSP, 256), 256, 0, stream>>>(
                    qdw + (size_t)h * c * NB + (size_t)b * NSP, NB, 1,
                    c, NSP, qA + (size_t)bh * c * NSP);
            }
        for (int bh = 0; bh < 8; ++bh)   // attn = Q (c x 4096) * K^T (4096 x 960)
            gemm_wmma<0><<<gB(c, KVC), 128, 0, stream>>>(
                qA + (size_t)bh * c * NSP, kB + (size_t)bh * NSP * KVC,
                attn + (size_t)bh * c * KVC, nullptr, c, KVC, NSP);
        inorm_stats<<<8, 256, 0, stream>>>(attn, stats, c);
        softmax_pack<<<8 * c, 256, 0, stream>>>(attn, stats, prA, c);
        for (int bh = 0; bh < 8; ++bh)   // O = P (c x 960) * V (960 x 4096)
            gemm_wmma<0><<<gB(c, NSP), 128, 0, stream>>>(
                prA + (size_t)bh * c * KVC, vB + (size_t)bh * KVC * NSP,
                obh + (size_t)bh * c * NSP, nullptr, c, NSP, KVC);
        head_mean<<<cdiv((size_t)c * NB, 256), 256, 0, stream>>>(obh, omean, c);
        pack_B_k<<<cdiv((size_t)c * NB, 256), 256, 0, stream>>>(omean, NB, 1, c, NB, omB);
        pack_A_k<<<cdiv((size_t)c * c, 256), 256, 0, stream>>>(prm(i, 4), c, 1, c, c, WpA);
        gemm_wmma<0><<<gB(c, NB), 128, 0, stream>>>(WpA, omB, oproj, nullptr, c, NB, c);
        residual_add<<<cdiv((size_t)2 * c * NSP, 256), 256, 0, stream>>>(emb[i], oproj, outp, c);

        // ---- FFN phase (scratch reset; cx lives in d_out) ----
        off = scratch0;
        const int dc = c / 4, hid = 4 * c;
        float* xln = (float*)alloc((size_t)c * NB * 4);
        float* x1  = (float*)alloc((size_t)dc * NB * 4);
        _Float16* xcB = (_Float16*)alloc((size_t)c * NB * 2);
        _Float16* W1A = (_Float16*)alloc((size_t)2 * hid * c * 2);
        float* t  = (float*)alloc((size_t)2 * hid * NB * 4);
        float* t1 = (float*)alloc((size_t)hid * NB * 4);
        float* u  = (float*)alloc((size_t)hid * NB * 4);
        _Float16* uB  = (_Float16*)alloc((size_t)hid * NB * 2);
        _Float16* W2A = (_Float16*)alloc((size_t)c * hid * 2);
        float* yb = (float*)alloc((size_t)c * NB * 4);
        float* gb = (float*)alloc((size_t)2 * c * 4);

        ln_branch<true, false><<<NB, 256, 0, stream>>>(outp, c, prm(i, 5), prm(i, 6),
                                                       xln, nullptr);
        pconv<<<cdiv((size_t)2 * dc * NSP, 256), 256, 0, stream>>>(xln, x1, prm(i, 7), c, dc);
        packB_ffn<<<cdiv((size_t)c * NB, 256), 256, 0, stream>>>(xln, x1, c, dc, xcB);
        pack_A_k<<<cdiv((size_t)2 * hid * c, 256), 256, 0, stream>>>(
            prm(i, 8), c, 1, 2 * hid, c, W1A);
        gemm_wmma<2><<<gB(2 * hid, NB), 128, 0, stream>>>(W1A, xcB, t, prm(i, 9),
                                                          2 * hid, NB, c);
        dwconv<3, true, false><<<cdiv((size_t)hid * NB, 256), 256, 0, stream>>>(
            t, t1, prm(i, 10), prm(i, 11), nullptr, hid);
        dwconv<5, true, true><<<cdiv((size_t)hid * NB, 256), 256, 0, stream>>>(
            t + (size_t)hid * NB, u, prm(i, 12), prm(i, 13), t1, hid);
        pack_B_k<<<cdiv((size_t)hid * NB, 256), 256, 0, stream>>>(u, NB, 1, hid, NB, uB);
        pack_A_k<<<cdiv((size_t)c * hid, 256), 256, 0, stream>>>(
            prm(i, 14), hid, 1, c, hid, W2A);
        gemm_wmma<1><<<gB(c, NB), 128, 0, stream>>>(W2A, uB, yb, prm(i, 15), c, NB, hid);
        col_mean<<<2 * c, 256, 0, stream>>>(yb, gb, c);
        eca_add_out<<<cdiv((size_t)2 * c * NSP, 256), 256, 0, stream>>>(
            yb, gb, prm(i, 16), outp, c);
    }
}
